// GNN_node_32684701122614
// MI455X (gfx1250) — matrix operand order repaired and checked by reading, not compiled, not run on hardware
//
#include <hip/hip_runtime.h>
#include <hip/hip_bf16.h>
#include <stdint.h>

#define N_INST 300000
#define N_NET  100000
#define N_TOT  400000
#define EMB    128
#define NEDGE  2000000

typedef __bf16 bf16_t;
typedef __attribute__((ext_vector_type(16))) __bf16 v16bf;
typedef __attribute__((ext_vector_type(8)))  __bf16 v8bf;
typedef __attribute__((ext_vector_type(4)))  __bf16 v4bf;
typedef __attribute__((ext_vector_type(8)))  float  v8f;

__device__ __forceinline__ float leakyf(float x){ return x > 0.f ? x : 0.1f*x; }

__device__ __forceinline__ bf16_t f2bf(float f){
  union { float f; uint32_t u; } x; x.f = f;
  uint32_t r = x.u + 0x7FFFu + ((x.u >> 16) & 1u);   // round-to-nearest-even
  uint16_t h = (uint16_t)(r >> 16);
  bf16_t o; __builtin_memcpy(&o, &h, 2); return o;
}

// ---------------- f32 -> bf16 bulk convert (8 elems / thread) ----------------
__global__ void k_cvt_bf16(const float* __restrict__ in, bf16_t* __restrict__ out, long nchunk){
  long i = (long)blockIdx.x*blockDim.x + threadIdx.x;
  if (i >= nchunk) return;
  const float4* p = (const float4*)(in + i*8);
  float4 a = p[0], b = p[1];
  v8bf o;
  o[0]=f2bf(a.x); o[1]=f2bf(a.y); o[2]=f2bf(a.z); o[3]=f2bf(a.w);
  o[4]=f2bf(b.x); o[5]=f2bf(b.y); o[6]=f2bf(b.z); o[7]=f2bf(b.w);
  *(v8bf*)(out + i*8) = o;
}

// ---------------- WMMA GEMM: out[M,Nout] = A[M,K] @ W[Nout,K]^T + bias -------
// Each wave computes a 16-row x 128-col strip: 1 A fragment, 8 WMMAs per k-step.
// mode 0: leaky -> bf16 (outB)
// mode 1: leaky -> f32 (outF) + bf16 mirror (outB)
// mode 2: conv  -> outR = relu(x) f32 ; outF = relu(x+root)*rdeg[row] f32
__global__ void k_gemm_wmma(const bf16_t* __restrict__ A, const bf16_t* __restrict__ W,
                            const float* __restrict__ bias, const float* __restrict__ root,
                            const float* __restrict__ rdeg,
                            float* __restrict__ outF, float* __restrict__ outR,
                            bf16_t* __restrict__ outB,
                            int M, int K, int Nout, int mode){
  const int lane = threadIdx.x & 31;
  const int wv   = threadIdx.x >> 5;
  const int m0   = blockIdx.x*128 + wv*16;        // 8 waves -> 8 row tiles
  if (m0 >= M) return;                             // wave-uniform: EXEC stays all-ones
  const int n0   = blockIdx.y * 128;
  const int half = lane >> 4;
  const int l15  = lane & 15;
  const bf16_t* arow  = A + (size_t)(m0 + l15) * K;
  const bf16_t* wbase = W + (size_t)(n0 + l15) * K + 16*half;

  v8f acc[8] = {};
  for (int k0 = 0; k0 < K; k0 += 32){
    // A 16x32 bf16 fragment: lane holds K=[k0+16h, +16)
    v16bf a = *(const v16bf*)(arow + k0 + 16*half);
    #pragma unroll
    for (int t=0;t<8;t++){
      // B 32x16 fragment for column tile t: lane holds col n0+16t+l15, K=[k0+16h,+16)
      v16bf b = *(const v16bf*)(wbase + (size_t)(t*16)*K + k0);
      acc[t] = __builtin_amdgcn_wmma_f32_16x16x32_bf16(false, a, false, b, (short)0, acc[t], false, false);
    }
  }

  #pragma unroll
  for (int t=0;t<8;t++){
    int nc = n0 + t*16 + l15;
    float bcol = bias[nc];
    float rcol = (mode == 2) ? root[nc] : 0.f;
    #pragma unroll
    for (int r=0;r<8;r++){
      int row = m0 + 8*half + r;                   // C/D layout: VGPR r -> row m0+8*half+r
      size_t idx = (size_t)row * Nout + nc;
      float v = acc[t][r] + bcol;
      if (mode == 0){
        outB[idx] = f2bf(leakyf(v));
      } else if (mode == 1){
        float y = leakyf(v);
        outF[idx] = y;
        outB[idx] = f2bf(y);
      } else {
        outR[idx] = fmaxf(v, 0.f);
        outF[idx] = fmaxf(v + rcol, 0.f) * rdeg[row];
      }
    }
  }
}

// ---------------- degree kernels ----------------
__global__ void k_deg_init(float* d1, float* d2, int n){
  int i = blockIdx.x*blockDim.x + threadIdx.x;
  if (i < n){ d1[i] = 1.0f; d2[i] = 1.0f; }
}
__global__ void k_deg_count(const int* __restrict__ r1, const int* __restrict__ r2,
                            float* d1, float* d2, int E){
  int e = blockIdx.x*blockDim.x + threadIdx.x;
  if (e >= E) return;
  unsafeAtomicAdd(&d1[r1[e]], 1.0f);
  unsafeAtomicAdd(&d2[r2[e]], 1.0f);
}
__global__ void k_deg_fin(float* deg1, float* deg2, float* rdeg1, float* rdeg2, int n){
  int i = blockIdx.x*blockDim.x + threadIdx.x;
  if (i >= n) return;
  float a = deg1[i]; rdeg1[i] = 1.0f/a; deg1[i] = rsqrtf(a);   // deg arrays become dinv
  float b = deg2[i]; rdeg2[i] = 1.0f/b; deg2[i] = rsqrtf(b);
}

// ---------------- edge scatter: one wave per edge, float4 per lane ----------------
__global__ void k_edge_agg(const int* __restrict__ row, const int* __restrict__ col,
                           const float* __restrict__ dinv,
                           const float* __restrict__ R, float* __restrict__ ACC, int E){
  int e = blockIdx.x*8 + (threadIdx.x>>5);
  if (e >= E) return;
  int lane = threadIdx.x & 31;
  int r = row[e], c = col[e];
  float nrm = dinv[r]*dinv[c];
  float4 v = *(const float4*)(R + (size_t)r*EMB + lane*4);
  float* dst = ACC + (size_t)c*EMB + lane*4;
  unsafeAtomicAdd(dst+0, nrm*v.x);
  unsafeAtomicAdd(dst+1, nrm*v.y);
  unsafeAtomicAdd(dst+2, nrm*v.z);
  unsafeAtomicAdd(dst+3, nrm*v.w);
}

// ------- residual + LayerNorm + leaky -> bf16 next-h + out slice, one wave/node ----
__global__ void k_post_ln(const float* __restrict__ ACC, const float* __restrict__ h0,
                          const float* __restrict__ g, const float* __restrict__ bb,
                          bf16_t* __restrict__ hnextb, float* __restrict__ out, int layer){
  int node = blockIdx.x*8 + (threadIdx.x>>5);     // N_TOT divisible by 8
  int lane = threadIdx.x & 31;
  size_t base = (size_t)node*EMB + lane*4;
  float4 a = *(const float4*)(ACC+base);
  float4 h = *(const float4*)(h0+base);
  float v0=a.x+h.x, v1=a.y+h.y, v2=a.z+h.z, v3=a.w+h.w;
  float s  = v0+v1+v2+v3;
  float s2 = v0*v0+v1*v1+v2*v2+v3*v3;
  #pragma unroll
  for (int off=16; off>0; off>>=1){
    s  += __shfl_xor(s,  off, 32);
    s2 += __shfl_xor(s2, off, 32);
  }
  float mean = s * (1.f/128.f);
  float var  = s2 * (1.f/128.f) - mean*mean;
  float inv  = rsqrtf(var + 1e-5f);
  int d = lane*4;
  float4 o;
  o.x = leakyf((v0-mean)*inv*g[d+0] + bb[d+0]);
  o.y = leakyf((v1-mean)*inv*g[d+1] + bb[d+1]);
  o.z = leakyf((v2-mean)*inv*g[d+2] + bb[d+2]);
  o.w = leakyf((v3-mean)*inv*g[d+3] + bb[d+3]);
  v4bf ob; ob[0]=f2bf(o.x); ob[1]=f2bf(o.y); ob[2]=f2bf(o.z); ob[3]=f2bf(o.w);
  *(v4bf*)(hnextb+base) = ob;                      // next layer's GEMM input (bf16)
  if (node >= N_INST){
    *(float4*)(out + (size_t)(node-N_INST)*512 + (size_t)(layer+1)*EMB + d) = o;
  }
}

// ---------------- h0 net-rows -> out[:, 0:128] ----------------
__global__ void k_copy_h0(const float* __restrict__ h0, float* __restrict__ out){
  long i = (long)blockIdx.x*blockDim.x + threadIdx.x;     // one float4 chunk each
  long total = (long)N_NET * 32;
  if (i >= total) return;
  long node = i >> 5; int c = (int)(i & 31) * 4;
  float4 v = *(const float4*)(h0 + (size_t)(N_INST+node)*EMB + c);
  *(float4*)(out + (size_t)node*512 + c) = v;
}

// ============================================================================
extern "C" void kernel_launch(void* const* d_in, const int* in_sizes, int n_in,
                              void* d_out, int out_size, void* d_ws, size_t ws_size,
                              hipStream_t stream) {
  const float* x      = (const float*)d_in[0];
  const float* xnet   = (const float*)d_in[1];
  const int*   ein    = (const int*)d_in[2];
  const int*   enn    = (const int*)d_in[3];
  const int* row1 = ein;  const int* col1 = ein + NEDGE;
  const int* row2 = enn;  const int* col2 = enn + NEDGE;
  const float* enc1W  = (const float*)d_in[5];
  const float* enc1b  = (const float*)d_in[6];
  const float* enc2W  = (const float*)d_in[7];
  const float* enc2b  = (const float*)d_in[8];
  const float* encn1W = (const float*)d_in[9];
  const float* encn1b = (const float*)d_in[10];
  const float* encn2W = (const float*)d_in[11];
  const float* encn2b = (const float*)d_in[12];
  const float* convW  = (const float*)d_in[13];
  const float* convb  = (const float*)d_in[14];
  const float* convr  = (const float*)d_in[15];
  const float* reconW = (const float*)d_in[16];
  const float* reconb = (const float*)d_in[17];
  const float* reconr = (const float*)d_in[18];
  const float* lng    = (const float*)d_in[19];
  const float* lnb    = (const float*)d_in[20];
  float* out = (float*)d_out;

  // ---- workspace layout ----
  char* ws = (char*)d_ws;
  const size_t szH = (size_t)N_TOT*EMB*sizeof(float);      // 204.8 MB
  const size_t szHB = (size_t)N_TOT*EMB*sizeof(bf16_t);    // 102.4 MB
  float*  h0  = (float*)(ws + 0*szH);
  float*  P0  = (float*)(ws + 1*szH);
  float*  P1  = (float*)(ws + 2*szH);
  float*  P2  = (float*)(ws + 3*szH);
  bf16_t* HBa = (bf16_t*)(ws + 4*szH);                     // conv1 / layer input (bf16)
  bf16_t* HBb = (bf16_t*)(ws + 4*szH + szHB);              // conv2 input (bf16)
  char* p = ws + 4*szH + 2*szHB;
  float* dinv1 = (float*)p; p += (size_t)N_TOT*4;
  float* rdeg1 = (float*)p; p += (size_t)N_TOT*4;
  float* dinv2 = (float*)p; p += (size_t)N_TOT*4;
  float* rdeg2 = (float*)p; p += (size_t)N_TOT*4;
  bf16_t* enc1Wb  = (bf16_t*)p;
  bf16_t* enc2Wb  = enc1Wb  + 256*64;
  bf16_t* encn1Wb = enc2Wb  + 128*256;
  bf16_t* encn2Wb = encn1Wb + 128*64;
  bf16_t* convWb  = encn2Wb + 128*128;
  bf16_t* reconWb = convWb  + 3*128*128;

  auto cvt = [&](const float* src, bf16_t* dst, long nelem){
    long nchunk = nelem / 8;
    k_cvt_bf16<<<(unsigned)((nchunk + 255)/256), 256, 0, stream>>>(src, dst, nchunk);
  };

  // ---- convert all weights to bf16 ----
  cvt(enc1W,  enc1Wb,  256*64);
  cvt(enc2W,  enc2Wb,  128*256);
  cvt(encn1W, encn1Wb, 128*64);
  cvt(encn2W, encn2Wb, 128*128);
  cvt(convW,  convWb,  3*128*128);
  cvt(reconW, reconWb, 3*128*128);

  // ---- degrees (row-index only; shared across layers) ----
  k_deg_init<<<(N_TOT+255)/256, 256, 0, stream>>>(dinv1, dinv2, N_TOT);
  k_deg_count<<<(NEDGE+255)/256, 256, 0, stream>>>(row1, row2, dinv1, dinv2, NEDGE);
  k_deg_fin<<<(N_TOT+255)/256, 256, 0, stream>>>(dinv1, dinv2, rdeg1, rdeg2, N_TOT);

  // ---- instance encoder: x[300k,64] -> 256 -> 128 ----
  cvt(x, HBa, (long)N_INST*64);
  k_gemm_wmma<<<dim3((N_INST+127)/128, 2), 256, 0, stream>>>(
      HBa, enc1Wb, enc1b, nullptr, nullptr, nullptr, nullptr, (bf16_t*)P1,
      N_INST, 64, 256, 0);
  k_gemm_wmma<<<dim3((N_INST+127)/128, 1), 256, 0, stream>>>(
      (bf16_t*)P1, enc2Wb, enc2b, nullptr, nullptr, h0, nullptr, HBa,
      N_INST, 256, 128, 1);

  // ---- net encoder: x_net[100k,64] -> 128 -> 128 ----
  cvt(xnet, HBb, (long)N_NET*64);
  k_gemm_wmma<<<dim3((N_NET+127)/128, 1), 256, 0, stream>>>(
      HBb, encn1Wb, encn1b, nullptr, nullptr, nullptr, nullptr, (bf16_t*)P2,
      N_NET, 64, 128, 0);
  k_gemm_wmma<<<dim3((N_NET+127)/128, 1), 256, 0, stream>>>(
      (bf16_t*)P2, encn2Wb, encn2b, nullptr, nullptr,
      h0 + (size_t)N_INST*EMB, nullptr, HBa + (size_t)N_INST*EMB,
      N_NET, 128, 128, 1);

  // ---- h0 net rows -> out columns [0,128) ----
  k_copy_h0<<<(unsigned)(((long)N_NET*32 + 255)/256), 256, 0, stream>>>(h0, out);

  // ---- 3 GNN layers ----
  for (int l = 0; l < 3; ++l){
    // conv (node->net edges): input HBa (bf16)
    k_gemm_wmma<<<dim3(N_TOT/128, 1), 256, 0, stream>>>(
        HBa, convWb + (size_t)l*128*128, convb + l*128, convr + l*128, rdeg1,
        /*outF=*/P2, /*outR=*/P1, nullptr, N_TOT, 128, 128, 2);
    k_edge_agg<<<NEDGE/8, 256, 0, stream>>>(row1, col1, dinv1, P1, P2, NEDGE);
    // re_conv (net->node edges): input = ACC (P2), produced by atomics -> convert
    cvt(P2, HBb, (long)N_TOT*EMB);
    k_gemm_wmma<<<dim3(N_TOT/128, 1), 256, 0, stream>>>(
        HBb, reconWb + (size_t)l*128*128, reconb + l*128, reconr + l*128, rdeg2,
        /*outF=*/P1, /*outR=*/P0, nullptr, N_TOT, 128, 128, 2);
    k_edge_agg<<<NEDGE/8, 256, 0, stream>>>(row2, col2, dinv2, P0, P1, NEDGE);
    // residual + LN + leaky -> bf16 next-h (HBa) and out slice
    k_post_ln<<<N_TOT/8, 256, 0, stream>>>(P1, h0, lng + l*128, lnb + l*128, HBa, out, l);
  }
}
